// Seq2Seq_65738769433296
// MI455X (gfx1250) — matrix operand order repaired and checked by reading, not compiled, not run on hardware
//
#include <hip/hip_runtime.h>
#include <hip/hip_bf16.h>

// ---------------------------------------------------------------------------
// Seq2Seq LSTM (H=1024) on gfx1250 / MI455X.
// bf16 weights (L2-resident: 96 MB total, halves the bandwidth-bound weight
// traffic), fp32 accumulation with V_WMMA_F32_16X16X32_BF16. Matvec-as-WMMA
// with the vector broadcast across the 16 N columns. Activations (x, h) are
// converted to bf16 ONCE per block into LDS so every B fragment is two
// ds_load_b128. Two independent accumulators (even/odd K-chunks) keep the
// matrix pipe from serializing on the C-accumulation chain.
// ---------------------------------------------------------------------------

#define H 1024
#define SRC_LEN 2048
#define TRG_LEN 512
#define V_OUT 32000

typedef __attribute__((ext_vector_type(16))) __bf16 v16bf;
typedef __attribute__((ext_vector_type(8)))  float  v8f;

// ---- fp32 -> bf16 (round-to-nearest-even, no NaN special-case) ------------
__device__ __forceinline__ unsigned f2bf(float f) {
  union { float f; unsigned u; } x; x.f = f;
  unsigned r = x.u + 0x7FFFu + ((x.u >> 16) & 1u);
  return r >> 16;
}

__device__ __forceinline__ float sigf(float x) { return 1.0f / (1.0f + __expf(-x)); }

// ---------------------------------------------------------------------------
// Weight packing: row-major fp32 [R][1024] -> per-(tile,kc,lane) contiguous
// 8-dword bf16 A-fragments in contiguous k order (k = 16h + s), not the ISA table order:
//   lane L, dword v holds K pair; K = kc*32 + v*2 + (L>>4)*16,
//   row = tile*16 + (L&15).
// Packed dword index d = ((tile*32 + kc)*32 + lane)*8 + v.
// ---------------------------------------------------------------------------
__global__ void pack_bf16_kernel(const float* __restrict__ W,
                                 unsigned* __restrict__ P, int ndw) {
  int d = blockIdx.x * 256 + threadIdx.x;
  if (d >= ndw) return;
  int v    =  d        & 7;
  int lane = (d >> 3)  & 31;
  int kc   = (d >> 8)  & 31;
  int tile =  d >> 13;
  int row  = tile * 16 + (lane & 15);
  int k    = kc * 32 + (v * 2) + ((lane >> 4) << 4);
  const float* wr = W + (long)row * H + k;
  P[d] = f2bf(wr[0]) | (f2bf(wr[1]) << 16);
}

__global__ void init_state_kernel(float* h0, float* h1, float* c, int* tok) {
  int t = threadIdx.x;
  h0[t] = 0.0f; h1[t] = 0.0f; c[t] = 0.0f;
  if (t == 0) *tok = 0;
}

// ---------------------------------------------------------------------------
// One fused LSTM step. Grid = 32 blocks x 256 threads (8 waves).
// Prologue: all threads convert x (= emb[row]) and h_in to bf16 in LDS.
// Wave w = (gate q = w>>1, sub s = w&1) computes 16 rows of
// g = W_ih@x + W_hh@h via 64 WMMAs split across two independent accumulator
// chains, stages results to LDS, then 32 threads apply biases + gate math
// and write h_out (double-buffered) and c.
// ---------------------------------------------------------------------------
__global__ void __launch_bounds__(256)
lstm_step_kernel(const unsigned* __restrict__ wih_p,
                 const unsigned* __restrict__ whh_p,
                 const float* __restrict__ emb,
                 const int* __restrict__ row_idx,
                 const float* __restrict__ b_ih,
                 const float* __restrict__ b_hh,
                 const float* __restrict__ h_in,
                 float* __restrict__ h_out,
                 float* __restrict__ c) {
  __shared__ __align__(32) unsigned short xb[H];
  __shared__ __align__(32) unsigned short hb[H];
  __shared__ float gbuf[4][32];

  const int tid = threadIdx.x;
  const int wv  = tid >> 5;
  const int ln  = tid & 31;
  const int q   = wv >> 1;
  const int s   = wv & 1;
  const int bb  = blockIdx.x;
  const int tile = q * 64 + bb * 2 + s;          // row-tile in 4096-row W
  const float* x = emb + (long)(*row_idx) * H;   // uniform scalar load

  // ---- activations -> bf16 in LDS (once per block) ----
#pragma unroll
  for (int i = tid; i < H; i += 256) {
    xb[i] = (unsigned short)f2bf(x[i]);
    hb[i] = (unsigned short)f2bf(h_in[i]);
  }
  __syncthreads();

  const int bco = (ln >> 4) << 4;                // B-frag K base per lane half

  v8f acc0 = {};
  v8f acc1 = {};
#pragma unroll 2
  for (int kc = 0; kc < 32; kc += 2) {
    v16bf a0 = *(const v16bf*)(wih_p + (((long)tile * 32 + kc)     * 32 + ln) * 8);
    v16bf a1 = *(const v16bf*)(wih_p + (((long)tile * 32 + kc + 1) * 32 + ln) * 8);
    v16bf b0 = *(const v16bf*)(&xb[kc * 32 + bco]);        // 2x ds_load_b128
    v16bf b1 = *(const v16bf*)(&xb[(kc + 1) * 32 + bco]);
    acc0 = __builtin_amdgcn_wmma_f32_16x16x32_bf16(false, a0, false, b0,
                                                   (short)0, acc0, false, false);
    acc1 = __builtin_amdgcn_wmma_f32_16x16x32_bf16(false, a1, false, b1,
                                                   (short)0, acc1, false, false);
  }
#pragma unroll 2
  for (int kc = 0; kc < 32; kc += 2) {
    v16bf a0 = *(const v16bf*)(whh_p + (((long)tile * 32 + kc)     * 32 + ln) * 8);
    v16bf a1 = *(const v16bf*)(whh_p + (((long)tile * 32 + kc + 1) * 32 + ln) * 8);
    v16bf b0 = *(const v16bf*)(&hb[kc * 32 + bco]);
    v16bf b1 = *(const v16bf*)(&hb[(kc + 1) * 32 + bco]);
    acc0 = __builtin_amdgcn_wmma_f32_16x16x32_bf16(false, a0, false, b0,
                                                   (short)0, acc0, false, false);
    acc1 = __builtin_amdgcn_wmma_f32_16x16x32_bf16(false, a1, false, b1,
                                                   (short)0, acc1, false, false);
  }
  v8f acc = acc0 + acc1;

  // D layout: lane n<16, vgpr p -> M=p ; lane n>=16, vgpr p -> M=p+8.
  // Columns are identical (B broadcast) so lanes 0 and 16 hold the 16 outputs.
  if ((ln & 15) == 0) {
    int gr = s * 16 + ((ln >> 4) << 3);
#pragma unroll
    for (int p = 0; p < 8; ++p) gbuf[q][gr + p] = acc[p];
  }
  __syncthreads();
  if (tid < 32) {
    int j = bb * 32 + tid;
    float gi = gbuf[0][tid] + b_ih[j]          + b_hh[j];
    float gf = gbuf[1][tid] + b_ih[H + j]      + b_hh[H + j];
    float gg = gbuf[2][tid] + b_ih[2 * H + j]  + b_hh[2 * H + j];
    float go = gbuf[3][tid] + b_ih[3 * H + j]  + b_hh[3 * H + j];
    float cn = sigf(gf) * c[j] + sigf(gi) * tanhf(gg);
    float hn = sigf(go) * tanhf(cn);
    c[j] = cn;
    h_out[j] = hn;
  }
}

// ---------------------------------------------------------------------------
// fc: z = fc_w @ h + fc_b.  2000 row-tiles; grid = 250 x 256 (1 tile / wave).
// ---------------------------------------------------------------------------
__global__ void __launch_bounds__(256)
fc_kernel(const unsigned* __restrict__ fc_p,
          const float* __restrict__ h,
          const float* __restrict__ fc_b,
          float* __restrict__ z) {
  __shared__ __align__(32) unsigned short hbv[H];
  const int tid = threadIdx.x;
  const int wv  = tid >> 5;
  const int ln  = tid & 31;
  const int tile = blockIdx.x * 8 + wv;          // 0..1999
  const int bco  = (ln >> 4) << 4;

#pragma unroll
  for (int i = tid; i < H; i += 256) hbv[i] = (unsigned short)f2bf(h[i]);
  __syncthreads();

  v8f acc0 = {};
  v8f acc1 = {};
#pragma unroll 2
  for (int kc = 0; kc < 32; kc += 2) {
    v16bf a0 = *(const v16bf*)(fc_p + (((long)tile * 32 + kc)     * 32 + ln) * 8);
    v16bf a1 = *(const v16bf*)(fc_p + (((long)tile * 32 + kc + 1) * 32 + ln) * 8);
    v16bf b0 = *(const v16bf*)(&hbv[kc * 32 + bco]);
    v16bf b1 = *(const v16bf*)(&hbv[(kc + 1) * 32 + bco]);
    acc0 = __builtin_amdgcn_wmma_f32_16x16x32_bf16(false, a0, false, b0,
                                                   (short)0, acc0, false, false);
    acc1 = __builtin_amdgcn_wmma_f32_16x16x32_bf16(false, a1, false, b1,
                                                   (short)0, acc1, false, false);
  }
  v8f acc = acc0 + acc1;

  if ((ln & 15) == 0) {
    int base = tile * 16 + ((ln >> 4) << 3);
#pragma unroll
    for (int p = 0; p < 8; ++p) z[base + p] = acc[p] + fc_b[base + p];
  }
}

// ---------------------------------------------------------------------------
// Single-block max / argmax (first occurrence) / logsumexp over z[32000].
// Writes lse and next token.
// ---------------------------------------------------------------------------
__global__ void __launch_bounds__(1024)
reduce_kernel(const float* __restrict__ z, float* __restrict__ lse,
              int* __restrict__ tok) {
  __shared__ float sv[1024];
  __shared__ int   si[1024];
  const int tid = threadIdx.x;
  float m = -3.402823466e38f; int mi = 0x7fffffff;
  for (int i = tid; i < V_OUT; i += 1024) {
    float v = z[i];
    if (v > m || (v == m && i < mi)) { m = v; mi = i; }
  }
  sv[tid] = m; si[tid] = mi;
  __syncthreads();
  for (int st = 512; st > 0; st >>= 1) {
    if (tid < st) {
      float ov = sv[tid + st]; int oi = si[tid + st];
      if (ov > sv[tid] || (ov == sv[tid] && oi < si[tid])) { sv[tid] = ov; si[tid] = oi; }
    }
    __syncthreads();
  }
  float gm = sv[0]; int gi = si[0];
  __syncthreads();
  float s = 0.0f;
  for (int i = tid; i < V_OUT; i += 1024) s += __expf(z[i] - gm);
  sv[tid] = s;
  __syncthreads();
  for (int st = 512; st > 0; st >>= 1) {
    if (tid < st) sv[tid] += sv[tid + st];
    __syncthreads();
  }
  if (tid == 0) { *lse = gm + __logf(sv[0]); *tok = gi; }
}

__global__ void logp_kernel(const float* __restrict__ z,
                            const float* __restrict__ lse,
                            float* __restrict__ out) {
  int i = blockIdx.x * 256 + threadIdx.x;
  if (i < V_OUT) out[i] = z[i] - *lse;
}

// ---------------------------------------------------------------------------
extern "C" void kernel_launch(void* const* d_in, const int* in_sizes, int n_in,
                              void* d_out, int out_size, void* d_ws, size_t ws_size,
                              hipStream_t stream) {
  (void)in_sizes; (void)n_in; (void)out_size; (void)ws_size;
  const int*   src      = (const int*)  d_in[0];
  const float* enc_emb  = (const float*)d_in[4];
  const float* enc_w_ih = (const float*)d_in[5];
  const float* enc_w_hh = (const float*)d_in[6];
  const float* enc_b_ih = (const float*)d_in[7];
  const float* enc_b_hh = (const float*)d_in[8];
  const float* dec_emb  = (const float*)d_in[9];
  const float* dec_w_ih = (const float*)d_in[10];
  const float* dec_w_hh = (const float*)d_in[11];
  const float* dec_b_ih = (const float*)d_in[12];
  const float* dec_b_hh = (const float*)d_in[13];
  const float* fc_w     = (const float*)d_in[14];
  const float* fc_b     = (const float*)d_in[15];
  float* out = (float*)d_out;

  // ---- workspace layout (bytes) ----
  char* w = (char*)d_ws;
  const size_t LSTM_W_B = (size_t)4 * H * H * 2;        // 8 MB bf16 packed
  const size_t FC_W_B   = (size_t)V_OUT * H * 2;        // 64 MB bf16 packed
  unsigned* enc_ih_p = (unsigned*)(w);
  unsigned* enc_hh_p = (unsigned*)(w + LSTM_W_B);
  unsigned* dec_ih_p = (unsigned*)(w + 2 * LSTM_W_B);
  unsigned* dec_hh_p = (unsigned*)(w + 3 * LSTM_W_B);
  unsigned* fc_p     = (unsigned*)(w + 4 * LSTM_W_B);
  float* h0  = (float*)(w + 4 * LSTM_W_B + FC_W_B);
  float* h1  = h0 + H;
  float* cst = h1 + H;
  float* z   = cst + H;
  float* lse = z + V_OUT;
  int*   tok = (int*)(lse + 1);

  // ---- prologue: pack weights to bf16 WMMA-fragment layout; zero state ----
  const int lstm_dw = 4 * H * H / 2;        // 2,097,152 dwords
  const int fc_dw   = V_OUT * H / 2;        // 16,384,000 dwords
  pack_bf16_kernel<<<lstm_dw / 256, 256, 0, stream>>>(enc_w_ih, enc_ih_p, lstm_dw);
  pack_bf16_kernel<<<lstm_dw / 256, 256, 0, stream>>>(enc_w_hh, enc_hh_p, lstm_dw);
  pack_bf16_kernel<<<lstm_dw / 256, 256, 0, stream>>>(dec_w_ih, dec_ih_p, lstm_dw);
  pack_bf16_kernel<<<lstm_dw / 256, 256, 0, stream>>>(dec_w_hh, dec_hh_p, lstm_dw);
  pack_bf16_kernel<<<fc_dw   / 256, 256, 0, stream>>>(fc_w,     fc_p,     fc_dw);
  init_state_kernel<<<1, 1024, 0, stream>>>(h0, h1, cst, tok);

  // ---- encoder: 2048 sequential fused LSTM steps (h double-buffered) ----
  long g = 0;
  for (int t = 0; t < SRC_LEN; ++t, ++g) {
    const float* hin = (g & 1) ? h1 : h0;
    float*       hot = (g & 1) ? h0 : h1;
    lstm_step_kernel<<<32, 256, 0, stream>>>(enc_ih_p, enc_hh_p, enc_emb,
                                             src + t, enc_b_ih, enc_b_hh,
                                             hin, hot, cst);
  }

  // ---- decoder: 512 steps of LSTM + fc + log_softmax + argmax feedback ----
  for (int t = 0; t < TRG_LEN; ++t, ++g) {
    const float* hin = (g & 1) ? h1 : h0;
    float*       hot = (g & 1) ? h0 : h1;
    lstm_step_kernel<<<32, 256, 0, stream>>>(dec_ih_p, dec_hh_p, dec_emb,
                                             tok, dec_b_ih, dec_b_hh,
                                             hin, hot, cst);
    fc_kernel<<<250, 256, 0, stream>>>(fc_p, hot, fc_b, z);
    reduce_kernel<<<1, 1024, 0, stream>>>(z, lse, tok);
    logp_kernel<<<125, 256, 0, stream>>>(z, lse, out + (size_t)t * V_OUT);
  }
}
